// BaseDenseAttention_37950331027620
// MI455X (gfx1250) — matrix operand order repaired and checked by reading, not compiled, not run on hardware
//
#include <hip/hip_runtime.h>

#define TQ 4096
#define TV 4096
#define DD 64
#define LOG2E 1.44269504088896340736f

typedef __attribute__((ext_vector_type(16))) __bf16 v16bf;
typedef __attribute__((ext_vector_type(2)))  __bf16 bf16x2;
typedef __attribute__((ext_vector_type(8)))  float  v8f;

union Frag { uint4 u[2]; v16bf v; };

// native f32 -> bf16 (backend picks v_cvt_pk_bf16_f32-class ops)
__device__ __forceinline__ unsigned short bfbits(float x) {
  __bf16 h = (__bf16)x;
  return __builtin_bit_cast(unsigned short, h);
}
__device__ __forceinline__ unsigned int cvt2(float a, float b) {
  bf16x2 v = { (__bf16)a, (__bf16)b };
  return __builtin_bit_cast(unsigned int, v);
}
__device__ __forceinline__ uint4 pack8(const float* __restrict__ s) {
  const float4* f = (const float4*)s;
  float4 a = f[0], b = f[1];
  uint4 r;
  r.x = cvt2(a.x, a.y); r.y = cvt2(a.z, a.w);
  r.z = cvt2(b.x, b.y); r.w = cvt2(b.z, b.w);
  return r;
}
__device__ __forceinline__ v8f wmma_bf16(v16bf a, v16bf b, v8f c) {
  return __builtin_amdgcn_wmma_f32_16x16x32_bf16(false, a, false, b, (short)0, c,
                                                 false, false);
}
__device__ __forceinline__ float rmax16(float x) {   // reduce within 16-lane half
  x = fmaxf(x, __shfl_xor(x, 1)); x = fmaxf(x, __shfl_xor(x, 2));
  x = fmaxf(x, __shfl_xor(x, 4)); x = fmaxf(x, __shfl_xor(x, 8));
  return x;
}
__device__ __forceinline__ float rsum16(float x) {
  x += __shfl_xor(x, 1); x += __shfl_xor(x, 2);
  x += __shfl_xor(x, 4); x += __shfl_xor(x, 8);
  return x;
}

// LDS pitches (halves): 16B-aligned rows, bank-conflict padding
#define VR_P 72   // [key][d]   32 x 64 used
#define VT_P 40   // [d][key]   64 x 32 used
#define PL_P 40   // per-wave P 16 x 32 used

__global__ __launch_bounds__(128) void flash_attn_causal(
    const float* __restrict__ Q, const float* __restrict__ V,
    const unsigned char* __restrict__ QM, const unsigned char* __restrict__ VM,
    float* __restrict__ O)
{
  __shared__ unsigned short VR[32 * VR_P];
  __shared__ unsigned short VT[64 * VT_P];
  __shared__ unsigned short PL[4][16 * PL_P];

  const int b    = blockIdx.y;
  const int wgq0 = blockIdx.x * 64;
  const int wave = threadIdx.x >> 5;
  const int lane = threadIdx.x & 31;
  const int hl   = lane >> 4;     // lane half (WMMA layout)
  const int ln   = lane & 15;
  const int q0w  = wgq0 + wave * 16;

  // ---- Q tile -> bf16 A-fragments (A layout: lane row = q0w+ln) ----
  Frag aq0, aq1;
  {
    const float* qr = Q + ((size_t)b * TQ + q0w + ln) * DD;
    aq0.u[0] = pack8(qr + hl * 16);           // K = hl*16 .. +7
    aq0.u[1] = pack8(qr + 8 + hl * 16);       // K = hl*16+8 .. +7
    aq1.u[0] = pack8(qr + 32 + hl * 16);
    aq1.u[1] = pack8(qr + 40 + hl * 16);
  }

  v8f o0 = {}, o1 = {}, o2 = {}, o3 = {};     // O accumulator, cols 0..63
  float m[8], l[8];
#pragma unroll
  for (int r = 0; r < 8; ++r) { m[r] = -1e30f; l[r] = 0.0f; }

  // ---- staging identity + register prefetch of V block 0 ----
  const int key  = threadIdx.x >> 2;                  // 0..31
  const int dseg = (threadIdx.x & 3) << 4;            // 0,16,32,48
  const float* vbase = V + ((size_t)b * TV + key) * DD + dseg;
  float4 vblk[4];
#pragma unroll
  for (int i = 0; i < 4; ++i) vblk[i] = ((const float4*)vbase)[i];

  const int nkv = (wgq0 >> 5) + 2;            // causal: kv blocks <= last row
  for (int it = 0; it < nkv; ++it) {
    const int kv0 = it << 5;

    // ---- convert the prefetched V block, store both LDS layouts ----
    {
      unsigned short t16[16];
#pragma unroll
      for (int i = 0; i < 4; ++i) {
        t16[i*4+0] = bfbits(vblk[i].x); t16[i*4+1] = bfbits(vblk[i].y);
        t16[i*4+2] = bfbits(vblk[i].z); t16[i*4+3] = bfbits(vblk[i].w);
      }
      uint4* dr = (uint4*)&VR[key * VR_P + dseg];
      unsigned int w0 = (unsigned int)t16[0]  | ((unsigned int)t16[1]  << 16);
      unsigned int w1 = (unsigned int)t16[2]  | ((unsigned int)t16[3]  << 16);
      unsigned int w2 = (unsigned int)t16[4]  | ((unsigned int)t16[5]  << 16);
      unsigned int w3 = (unsigned int)t16[6]  | ((unsigned int)t16[7]  << 16);
      unsigned int w4 = (unsigned int)t16[8]  | ((unsigned int)t16[9]  << 16);
      unsigned int w5 = (unsigned int)t16[10] | ((unsigned int)t16[11] << 16);
      unsigned int w6 = (unsigned int)t16[12] | ((unsigned int)t16[13] << 16);
      unsigned int w7 = (unsigned int)t16[14] | ((unsigned int)t16[15] << 16);
      dr[0] = make_uint4(w0, w1, w2, w3);
      dr[1] = make_uint4(w4, w5, w6, w7);
#pragma unroll
      for (int i = 0; i < 16; ++i) VT[(dseg + i) * VT_P + key] = t16[i];
    }
    __syncthreads();

    // ---- issue global loads for next V block NOW; they overlap the WMMAs ----
    if (it + 1 < nkv) {
      const float* src = vbase + (size_t)(it + 1) * 32 * DD;
#pragma unroll
      for (int i = 0; i < 4; ++i) vblk[i] = ((const float4*)src)[i];
      if (it + 2 < nkv) __builtin_prefetch(src + 32 * DD, 0, 0);
    }

    if (kv0 <= q0w + 15) {                    // wave has unmasked work here
      // ---- gemm1: S(16x32) = Q(16x64) @ K^T, two 16-key tiles ----
      v8f s0 = {}, s1 = {};
      {
        Frag bf; const uint4* p;
        p = (const uint4*)&VR[(0  + ln) * VR_P + 0  + hl * 16];
        bf.u[0] = p[0]; bf.u[1] = p[1];
        s0 = wmma_bf16(aq0.v, bf.v, s0);
        p = (const uint4*)&VR[(0  + ln) * VR_P + 32 + hl * 16];
        bf.u[0] = p[0]; bf.u[1] = p[1];
        s0 = wmma_bf16(aq1.v, bf.v, s0);
        p = (const uint4*)&VR[(16 + ln) * VR_P + 0  + hl * 16];
        bf.u[0] = p[0]; bf.u[1] = p[1];
        s1 = wmma_bf16(aq0.v, bf.v, s1);
        p = (const uint4*)&VR[(16 + ln) * VR_P + 32 + hl * 16];
        bf.u[0] = p[0]; bf.u[1] = p[1];
        s1 = wmma_bf16(aq1.v, bf.v, s1);
      }

      // ---- mask + online softmax (C layout: lane col = ln, rows hl*8+r) ----
      const int col0 = kv0 + ln, col1 = kv0 + 16 + ln;
      const unsigned char vm0 = VM[(size_t)b * TV + col0];
      const unsigned char vm1 = VM[(size_t)b * TV + col1];
      float p0[8], p1[8];
#pragma unroll
      for (int r = 0; r < 8; ++r) {
        const int row = q0w + hl * 8 + r;
        float s0r = s0[r] - (((col0 > row) || (vm0 == 0)) ? 1e9f : 0.0f);
        float s1r = s1[r] - (((col1 > row) || (vm1 == 0)) ? 1e9f : 0.0f);
        float mx = rmax16(fmaxf(s0r, s1r));
        float mn = fmaxf(m[r], mx);
        float sc = exp2f((m[r] - mn) * LOG2E);
        p0[r] = exp2f((s0r - mn) * LOG2E);
        p1[r] = exp2f((s1r - mn) * LOG2E);
        float rs = rsum16(p0[r] + p1[r]);
        l[r] = l[r] * sc + rs;
        m[r] = mn;
        o0[r] *= sc; o1[r] *= sc; o2[r] *= sc; o3[r] *= sc;
      }

      // ---- C layout -> A layout for P via per-wave LDS round trip ----
      unsigned short* pw = PL[wave];
#pragma unroll
      for (int r = 0; r < 8; ++r) {
        const int row = hl * 8 + r;
        pw[row * PL_P + ln]      = bfbits(p0[r]);
        pw[row * PL_P + 16 + ln] = bfbits(p1[r]);
      }
      asm volatile("s_wait_dscnt 0" ::: "memory");
      Frag pa;
      pa.u[0] = *(const uint4*)&pw[ln * PL_P + hl * 16];
      pa.u[1] = *(const uint4*)&pw[ln * PL_P + 8 + hl * 16];

      // ---- gemm2: O(16x64) += P(16x32) @ V(32x64), B from transposed LDS ----
      {
        Frag bv; const uint4* p;
        p = (const uint4*)&VT[(0*16 + ln) * VT_P + hl * 16];
        bv.u[0] = p[0]; bv.u[1] = p[1];
        o0 = wmma_bf16(pa.v, bv.v, o0);
        p = (const uint4*)&VT[(1*16 + ln) * VT_P + hl * 16];
        bv.u[0] = p[0]; bv.u[1] = p[1];
        o1 = wmma_bf16(pa.v, bv.v, o1);
        p = (const uint4*)&VT[(2*16 + ln) * VT_P + hl * 16];
        bv.u[0] = p[0]; bv.u[1] = p[1];
        o2 = wmma_bf16(pa.v, bv.v, o2);
        p = (const uint4*)&VT[(3*16 + ln) * VT_P + hl * 16];
        bv.u[0] = p[0]; bv.u[1] = p[1];
        o3 = wmma_bf16(pa.v, bv.v, o3);
      }
    }
    __syncthreads();
  }

  // ---- epilogue: normalize by l, apply q_mask, coalesced stores ----
#pragma unroll
  for (int r = 0; r < 8; ++r) {
    const int row  = q0w + hl * 8 + r;
    const float qm = QM[(size_t)b * TQ + row] ? 1.0f : 0.0f;
    const float inv = qm / l[r];
    float* orow = O + ((size_t)b * TQ + row) * DD + ln;
    orow[0]  = o0[r] * inv;
    orow[16] = o1[r] * inv;
    orow[32] = o2[r] * inv;
    orow[48] = o3[r] * inv;
  }
}

extern "C" void kernel_launch(void* const* d_in, const int* in_sizes, int n_in,
                              void* d_out, int out_size, void* d_ws, size_t ws_size,
                              hipStream_t stream) {
  const float* Q = (const float*)d_in[0];
  const float* V = (const float*)d_in[1];
  const unsigned char* QM = (const unsigned char*)d_in[2];  // bool8 q_mask
  const unsigned char* VM = (const unsigned char*)d_in[3];  // bool8 v_mask
  float* O = (float*)d_out;
  dim3 grid(TQ / 64, 4);   // (q blocks of 64 rows, batch)
  flash_attn_causal<<<grid, dim3(128), 0, stream>>>(Q, V, QM, VM, O);
}
